// ReformerModelWrapper_47364899340315
// MI455X (gfx1250) — matrix operand order repaired and checked by reading, not compiled, not run on hardware
//
#include <hip/hip_runtime.h>
#include <hip/hip_bf16.h>
#include <math.h>

typedef __attribute__((ext_vector_type(16))) _Float16 v16h;
typedef __attribute__((ext_vector_type(8)))  _Float16 v8h;
typedef __attribute__((ext_vector_type(4)))  _Float16 v4h;
typedef __attribute__((ext_vector_type(8)))  float    v8f;

#define DMODEL 64
#define LSEQ   1280
#define NBATCH 16
#define NHEADS 4
#define DH     16
#define BHDIM  (NBATCH*NHEADS)   // 64
#define NHASH  4
#define BSZ    64
#define NBKT   20
#define NCHUNK (NHASH*NBKT)      // 80
#define NTOK   (NBATCH*LSEQ)     // 20480
#define NHL    (NHASH*LSEQ)      // 5120
#define FFDIM  256

__device__ __forceinline__ v16h cat8(v8h lo, v8h hi) {
  return __builtin_shufflevector(lo, hi, 0, 1, 2, 3, 4, 5, 6, 7,
                                 8, 9, 10, 11, 12, 13, 14, 15);
}
__device__ __forceinline__ v8h zero8() {
  v8h z;
#pragma unroll
  for (int h = 0; h < 8; ++h) z[h] = (_Float16)0.f;
  return z;
}

// ---------------------------------------------------------------------------
// Embedding: pad wave (B,2,1250) -> (B,1280,2), x = x@in_w + in_b, dup to x1/x2
// ---------------------------------------------------------------------------
__global__ void embed_kernel(const float* __restrict__ wave,
                             const float* __restrict__ in_w,
                             const float* __restrict__ in_b,
                             float* __restrict__ x1, float* __restrict__ x2) {
  int idx = blockIdx.x * blockDim.x + threadIdx.x;   // B*L*D
  if (idx >= NBATCH * LSEQ * DMODEL) return;
  int d = idx & (DMODEL - 1);
  int l = (idx >> 6) % LSEQ;
  int b = idx / (LSEQ * DMODEL);
  float v = in_b[d];
  if (l < 1250) {
    v += wave[(size_t)(b * 2 + 0) * 1250 + l] * in_w[0 * DMODEL + d]
       + wave[(size_t)(b * 2 + 1) * 1250 + l] * in_w[1 * DMODEL + d];
  }
  x1[idx] = v;
  x2[idx] = v;
}

// ---------------------------------------------------------------------------
// LayerNorm over D=64, one wave (32 lanes, 2 elems each) per token
// ---------------------------------------------------------------------------
__global__ void ln_kernel(const float* __restrict__ x,
                          const float* __restrict__ g,
                          const float* __restrict__ bta,
                          float* __restrict__ y) {
  int token = blockIdx.x * (blockDim.x >> 5) + (threadIdx.x >> 5);
  int lane = threadIdx.x & 31;
  if (token >= NTOK) return;
  const float* row = x + (size_t)token * DMODEL;
  float a0 = row[lane], a1 = row[lane + 32];
  float sum = a0 + a1;
  for (int off = 16; off; off >>= 1) sum += __shfl_down(sum, off);
  sum = __shfl(sum, 0);
  float mean = sum * (1.0f / 64.0f);
  float d0 = a0 - mean, d1 = a1 - mean;
  float vs = d0 * d0 + d1 * d1;
  for (int off = 16; off; off >>= 1) vs += __shfl_down(vs, off);
  vs = __shfl(vs, 0);
  float rstd = rsqrtf(vs * (1.0f / 64.0f) + 1e-5f);
  y[(size_t)token * DMODEL + lane]      = d0 * rstd * g[lane]      + bta[lane];
  y[(size_t)token * DMODEL + lane + 32] = d1 * rstd * g[lane + 32] + bta[lane + 32];
}

// ---------------------------------------------------------------------------
// WMMA GEMM: C[M x N] = A[M x KDIM] @ W[KDIM x N] (+bias)(+gelu)(+residual)
// Block = 128 threads (4 waves) -> 64 rows x 16 cols. B tile kept transposed
// in LDS so both WMMA operand fills are contiguous ds_load_b128s.
// ---------------------------------------------------------------------------
template <int KDIM, bool BIAS, bool GELU, bool RES>
__global__ void gemm_wmma(const float* __restrict__ A,
                          const float* __restrict__ W,
                          const float* __restrict__ bias,
                          const float* __restrict__ res,
                          float* __restrict__ C, int N) {
  __shared__ alignas(16) _Float16 As[64 * KDIM];     // row-major [row][k]
  __shared__ alignas(16) _Float16 WsT[16 * KDIM];    // transposed [n][k]
  int row0 = blockIdx.x * 64;
  int col0 = blockIdx.y * 16;
  int tid = threadIdx.x;

  // A tile: float4 global loads -> v4h LDS stores
  for (int i = tid; i < 64 * KDIM / 4; i += 128) {
    int r = i / (KDIM / 4), k4 = i % (KDIM / 4);
    float4 f = *(const float4*)(A + (size_t)(row0 + r) * KDIM + k4 * 4);
    v4h hv;
    hv[0] = (_Float16)f.x; hv[1] = (_Float16)f.y;
    hv[2] = (_Float16)f.z; hv[3] = (_Float16)f.w;
    *(v4h*)(As + r * KDIM + k4 * 4) = hv;
  }
  // W tile transposed: coalesced global reads, scattered LDS writes
  for (int i = tid; i < KDIM * 16; i += 128) {
    int k = i >> 4, n = i & 15;
    WsT[n * KDIM + k] = (_Float16)W[(size_t)k * N + (col0 + n)];
  }
  __syncthreads();

  int wv = tid >> 5, lane = tid & 31;
  int mrow = wv * 16;
  int m = lane & 15;
  int kb = (lane >> 4) * 8;    // A: K sub-block offset within 16
  int kg = (lane >> 4) * 16;   // unused
  const _Float16* Arow = As + (mrow + m) * KDIM;
  const _Float16* Brow = WsT + m * KDIM;

  v8f acc = {0.f, 0.f, 0.f, 0.f, 0.f, 0.f, 0.f, 0.f};
#pragma unroll
  for (int k0 = 0; k0 < KDIM; k0 += 32) {
    v8h alo = *(const v8h*)(Arow + k0 + kb);
    v8h ahi = *(const v8h*)(Arow + k0 + 16 + kb);
    v16h a = cat8(alo, ahi);
    v8h blo = *(const v8h*)(Brow + k0 + kb);
    v8h bhi = *(const v8h*)(Brow + k0 + 16 + kb);
    v16h bm = cat8(blo, bhi);
    acc = __builtin_amdgcn_wmma_f32_16x16x32_f16(false, a, false, bm,
                                                 (short)0, acc, false, false);
  }

  int mb = (lane >> 4) * 8;
#pragma unroll
  for (int r = 0; r < 8; ++r) {
    int mrowg = row0 + mrow + mb + r;
    float val = acc[r];
    if (BIAS) val += bias[col0 + m];
    if (GELU) val = 0.5f * val * (1.0f + erff(val * 0.70710678118654752f));
    size_t idx = (size_t)mrowg * N + col0 + m;
    if (RES) val += res[idx];
    C[idx] = val;
  }
}

// ---------------------------------------------------------------------------
// LSH hashing: buckets[bh][h*L + pos] = argmax([r, -r]) + h*NBKT
// rot staged in LDS (640 floats) once per block.
// ---------------------------------------------------------------------------
__global__ void hash_kernel(const float* __restrict__ qk,
                            const float* __restrict__ rot,   // (DH, NHASH, NBKT/2)
                            int* __restrict__ buckets) {
  __shared__ float rotS[DH * NHASH * (NBKT / 2)];
  int tid = threadIdx.x;
  for (int i = tid; i < DH * NHASH * (NBKT / 2); i += blockDim.x) rotS[i] = rot[i];
  __syncthreads();

  int idx = blockIdx.x * blockDim.x + tid;   // BH*L
  if (idx >= BHDIM * LSEQ) return;
  int l = idx % LSEQ;
  int bh = idx / LSEQ;
  int b = bh >> 2, head = bh & 3;
  const float4* q4 = (const float4*)(qk + ((size_t)(b * LSEQ + l)) * DMODEL + head * DH);
  float qr[DH];
  {
    float4 f0 = q4[0], f1 = q4[1], f2 = q4[2], f3 = q4[3];
    qr[0] = f0.x; qr[1] = f0.y; qr[2] = f0.z; qr[3] = f0.w;
    qr[4] = f1.x; qr[5] = f1.y; qr[6] = f1.z; qr[7] = f1.w;
    qr[8] = f2.x; qr[9] = f2.y; qr[10] = f2.z; qr[11] = f2.w;
    qr[12] = f3.x; qr[13] = f3.y; qr[14] = f3.z; qr[15] = f3.w;
  }
  for (int h = 0; h < NHASH; ++h) {
    float rv[NBKT / 2];
    for (int i = 0; i < NBKT / 2; ++i) {
      float r = 0.f;
#pragma unroll
      for (int f = 0; f < DH; ++f) r += qr[f] * rotS[(f * NHASH + h) * (NBKT / 2) + i];
      rv[i] = r;
    }
    float best = rv[0]; int arg = 0;
    for (int i = 1; i < NBKT / 2; ++i) if (rv[i] > best) { best = rv[i]; arg = i; }
    for (int i = 0; i < NBKT / 2; ++i) if (-rv[i] > best) { best = -rv[i]; arg = i + NBKT / 2; }
    buckets[(size_t)bh * NHL + h * LSEQ + l] = arg + h * NBKT;
  }
}

// ---------------------------------------------------------------------------
// Stable counting sort by (bucket, position): sticker[rank] = ticker
// ---------------------------------------------------------------------------
__global__ void sort_kernel(const int* __restrict__ buckets,
                            int* __restrict__ sticker) {
  int bh = blockIdx.x;
  __shared__ int hist[NCHUNK];
  __shared__ int offs[NCHUNK];
  int tid = threadIdx.x;
  if (tid < NCHUNK) hist[tid] = 0;
  __syncthreads();
  const int* bk = buckets + (size_t)bh * NHL;
  for (int t = tid; t < NHL; t += 128) atomicAdd(&hist[bk[t]], 1);
  __syncthreads();
  if (tid == 0) {
    int acc = 0;
    for (int i = 0; i < NCHUNK; ++i) { offs[i] = acc; acc += hist[i]; }
  }
  __syncthreads();
  if (tid < NCHUNK) {
    int o = offs[tid];
    for (int t = 0; t < NHL; ++t)
      if (bk[t] == tid) { sticker[(size_t)bh * NHL + o] = t; ++o; }
  }
}

// ---------------------------------------------------------------------------
// Chunked LSH attention. Block per (bh, chunk): 128 threads = 4 waves.
// Key tile padded to K=32 with stored zeros -> all WMMA operand fills are
// plain contiguous ds_load_b128s (no selects, no exec masking).
// Softmax: single-exp pass, 1/sum deferred into the probs@V epilogue.
// ---------------------------------------------------------------------------
__global__ void attn_kernel(const float* __restrict__ qk,
                            const float* __restrict__ vv,
                            const int* __restrict__ sticker,
                            float* __restrict__ orounds,
                            float* __restrict__ lserounds) {
  __shared__ alignas(16) _Float16 sq[BSZ][DH];          // queries [row][d]
  __shared__ alignas(16) _Float16 sk[2 * BSZ][32];      // keys [key][d], d 16..31 = 0
  __shared__ alignas(16) _Float16 svT[DH][2 * BSZ];     // values transposed [d][key]
  __shared__ int qpos[BSZ];
  __shared__ int kpos[2 * BSZ];
  __shared__ int qtick[BSZ];
  __shared__ alignas(16) float dotsS[BSZ * 2 * BSZ];
  __shared__ alignas(16) _Float16 probsS[BSZ * 2 * BSZ];
  __shared__ float rowScale[BSZ];

  int bh = blockIdx.x, c = blockIdx.y;
  int b = bh >> 2, head = bh & 3;
  int tid = threadIdx.x;
  const int* stk = sticker + (size_t)bh * NHL;
  v8h z8 = zero8();

  if (tid < BSZ) {
    int t = stk[c * BSZ + tid];
    qtick[tid] = t;
    int pos = t % LSEQ;
    qpos[tid] = pos;
    const float4* qp4 = (const float4*)(qk + ((size_t)(b * LSEQ + pos)) * DMODEL + head * DH);
    float4 f0 = qp4[0], f1 = qp4[1], f2 = qp4[2], f3 = qp4[3];
    v8h q0, q1;
    q0[0] = (_Float16)f0.x; q0[1] = (_Float16)f0.y; q0[2] = (_Float16)f0.z; q0[3] = (_Float16)f0.w;
    q0[4] = (_Float16)f1.x; q0[5] = (_Float16)f1.y; q0[6] = (_Float16)f1.z; q0[7] = (_Float16)f1.w;
    q1[0] = (_Float16)f2.x; q1[1] = (_Float16)f2.y; q1[2] = (_Float16)f2.z; q1[3] = (_Float16)f2.w;
    q1[4] = (_Float16)f3.x; q1[5] = (_Float16)f3.y; q1[6] = (_Float16)f3.z; q1[7] = (_Float16)f3.w;
    *(v8h*)(&sq[tid][0]) = q0;
    *(v8h*)(&sq[tid][8]) = q1;
  }
  {
    int jj = tid;                               // 0..127 keys (this chunk + prev)
    int cc = (jj < BSZ) ? c : (c + NCHUNK - 1) % NCHUNK;
    int t = stk[cc * BSZ + (jj & (BSZ - 1))];
    int pos = t % LSEQ;
    kpos[jj] = pos;
    const float4* kp4 = (const float4*)(qk + ((size_t)(b * LSEQ + pos)) * DMODEL + head * DH);
    float kr[DH];
    {
      float4 f0 = kp4[0], f1 = kp4[1], f2 = kp4[2], f3 = kp4[3];
      kr[0] = f0.x; kr[1] = f0.y; kr[2] = f0.z; kr[3] = f0.w;
      kr[4] = f1.x; kr[5] = f1.y; kr[6] = f1.z; kr[7] = f1.w;
      kr[8] = f2.x; kr[9] = f2.y; kr[10] = f2.z; kr[11] = f2.w;
      kr[12] = f3.x; kr[13] = f3.y; kr[14] = f3.z; kr[15] = f3.w;
    }
    float ss = 0.f;
#pragma unroll
    for (int f = 0; f < DH; ++f) ss += kr[f] * kr[f];
    float inv = 1.0f / fmaxf(sqrtf(ss), 1e-12f);
    v8h n0, n1;
#pragma unroll
    for (int f = 0; f < 8; ++f) { n0[f] = (_Float16)(kr[f] * inv); n1[f] = (_Float16)(kr[f + 8] * inv); }
    *(v8h*)(&sk[jj][0])  = n0;
    *(v8h*)(&sk[jj][8])  = n1;
    *(v8h*)(&sk[jj][16]) = z8;                  // stored zero pad (K 16..31)
    *(v8h*)(&sk[jj][24]) = z8;
    const float4* vp4 = (const float4*)(vv + ((size_t)(b * LSEQ + pos)) * DMODEL + head * DH);
    float4 g0 = vp4[0], g1 = vp4[1], g2 = vp4[2], g3 = vp4[3];
    float vr[DH];
    vr[0] = g0.x; vr[1] = g0.y; vr[2] = g0.z; vr[3] = g0.w;
    vr[4] = g1.x; vr[5] = g1.y; vr[6] = g1.z; vr[7] = g1.w;
    vr[8] = g2.x; vr[9] = g2.y; vr[10] = g2.z; vr[11] = g2.w;
    vr[12] = g3.x; vr[13] = g3.y; vr[14] = g3.z; vr[15] = g3.w;
#pragma unroll
    for (int f = 0; f < DH; ++f) svT[f][jj] = (_Float16)vr[f];
  }
  __syncthreads();

  int wv = tid >> 5, lane = tid & 31;
  int m = lane & 15;
  int kb = (lane >> 4) * 8;
  int kg = (lane >> 4) * 16;

  // hoisted per-lane row metadata (8 output rows per lane)
  int qp8[8], qt8[8];
#pragma unroll
  for (int r = 0; r < 8; ++r) {
    int qm = wv * 16 + (lane >> 4) * 8 + r;
    qp8[r] = qpos[qm];
    qt8[r] = qtick[qm];
  }

  // ---- dots = Q @ K^T  (dh=16 zero-padded to K=32; scale 0.25; mask -5e4) ----
  {
    v8h alo = *(const v8h*)(&sq[wv * 16 + m][kb]);   // K 0..15 (per lane half)
    v16h a = cat8(alo, z8);                          // K 16..31 = 0
#pragma unroll
    for (int kt = 0; kt < 8; ++kt) {
      const _Float16* krow = &sk[kt * 16 + m][kb];   // elems 8..15 read stored zeros
      v8h blo = *(const v8h*)(krow);
      v8h bhi = *(const v8h*)(krow + 16);
      v16h bm = cat8(blo, bhi);
      v8f acc = {0.f, 0.f, 0.f, 0.f, 0.f, 0.f, 0.f, 0.f};
      acc = __builtin_amdgcn_wmma_f32_16x16x32_f16(false, a, false, bm,
                                                   (short)0, acc, false, false);
      int kn = kt * 16 + m;
      int kp = kpos[kn];
#pragma unroll
      for (int r = 0; r < 8; ++r) {
        int qm = wv * 16 + (lane >> 4) * 8 + r;
        float dv = acc[r] * 0.25f;
        if (qp8[r] == kp) dv = -5e4f;
        dotsS[qm * (2 * BSZ) + kn] = dv;
      }
    }
  }
  __syncthreads();

  // ---- row softmax: single exp pass; 1/sum deferred to epilogue ----
  if (tid < BSZ) {
    int row = tid;
    const float4* dp = (const float4*)(dotsS + row * (2 * BSZ));
    float mx = -3.0e38f;
#pragma unroll
    for (int k = 0; k < 32; ++k) {
      float4 v = dp[k];
      mx = fmaxf(mx, fmaxf(fmaxf(v.x, v.y), fmaxf(v.z, v.w)));
    }
    float sum = 0.f;
#pragma unroll
    for (int k = 0; k < 32; ++k) {
      float4 v = dp[k];
      float e0 = __expf(v.x - mx), e1 = __expf(v.y - mx);
      float e2 = __expf(v.z - mx), e3 = __expf(v.w - mx);
      sum += (e0 + e1) + (e2 + e3);
      v4h pv;
      pv[0] = (_Float16)e0; pv[1] = (_Float16)e1;
      pv[2] = (_Float16)e2; pv[3] = (_Float16)e3;
      *(v4h*)(probsS + row * (2 * BSZ) + k * 4) = pv;
    }
    rowScale[row] = 1.0f / sum;
    lserounds[(size_t)bh * NHL + qtick[row]] = mx + __logf(sum);
  }
  __syncthreads();

  // ---- out = (probsUnnorm @ V) * rowScale ----
  {
    const _Float16* prow = probsS + (wv * 16 + m) * (2 * BSZ);
    const _Float16* vrow = svT[m];                  // svT[n][*], n = lane&15
    v8f acc = {0.f, 0.f, 0.f, 0.f, 0.f, 0.f, 0.f, 0.f};
#pragma unroll
    for (int k0 = 0; k0 < 2 * BSZ; k0 += 32) {
      v8h alo = *(const v8h*)(prow + k0 + kb);
      v8h ahi = *(const v8h*)(prow + k0 + 16 + kb);
      v16h a = cat8(alo, ahi);
      v8h blo = *(const v8h*)(vrow + k0 + kb);
      v8h bhi = *(const v8h*)(vrow + k0 + 16 + kb);
      v16h bm = cat8(blo, bhi);
      acc = __builtin_amdgcn_wmma_f32_16x16x32_f16(false, a, false, bm,
                                                   (short)0, acc, false, false);
    }
    float sc8[8];
#pragma unroll
    for (int r = 0; r < 8; ++r)
      sc8[r] = rowScale[wv * 16 + (lane >> 4) * 8 + r];
#pragma unroll
    for (int r = 0; r < 8; ++r) {
      orounds[((size_t)bh * NHL + qt8[r]) * DH + m] = acc[r] * sc8[r];
    }
  }
}

// ---------------------------------------------------------------------------
// Combine hash rounds by softmax(LSE) weights; merge heads into (B, L, D)
// ---------------------------------------------------------------------------
__global__ void combine_kernel(const float* __restrict__ orounds,
                               const float* __restrict__ lserounds,
                               float* __restrict__ ctx) {
  int idx = blockIdx.x * blockDim.x + threadIdx.x;  // BH*L
  if (idx >= BHDIM * LSEQ) return;
  int l = idx % LSEQ;
  int bh = idx / LSEQ;
  int b = bh >> 2, head = bh & 3;
  float ls[NHASH], mx = -3.0e38f;
#pragma unroll
  for (int h = 0; h < NHASH; ++h) {
    ls[h] = lserounds[(size_t)bh * NHL + h * LSEQ + l];
    mx = fmaxf(mx, ls[h]);
  }
  float wsum = 0.f, wgt[NHASH];
#pragma unroll
  for (int h = 0; h < NHASH; ++h) { wgt[h] = __expf(ls[h] - mx); wsum += wgt[h]; }
  float inv = 1.0f / wsum;
#pragma unroll
  for (int d = 0; d < DH; ++d) {
    float o = 0.f;
#pragma unroll
    for (int h = 0; h < NHASH; ++h)
      o += orounds[((size_t)bh * NHL + h * LSEQ + l) * DH + d] * wgt[h];
    ctx[((size_t)(b * LSEQ + l)) * DMODEL + head * DH + d] = o * inv;
  }
}

// ---------------------------------------------------------------------------
// y = (0.5*(x1+x2)) @ out_w + out_b, trimmed to 1250
// ---------------------------------------------------------------------------
__global__ void final_kernel(const float* __restrict__ x1,
                             const float* __restrict__ x2,
                             const float* __restrict__ out_w,
                             const float* __restrict__ out_b,
                             float* __restrict__ y) {
  int idx = blockIdx.x * blockDim.x + threadIdx.x;  // B*1250
  if (idx >= NBATCH * 1250) return;
  int l = idx % 1250;
  int b = idx / 1250;
  const float* r1 = x1 + ((size_t)(b * LSEQ + l)) * DMODEL;
  const float* r2 = x2 + ((size_t)(b * LSEQ + l)) * DMODEL;
  float s = out_b[0];
#pragma unroll
  for (int d = 0; d < DMODEL; ++d) s += 0.5f * (r1[d] + r2[d]) * out_w[d];
  y[idx] = s;
}

// ---------------------------------------------------------------------------
extern "C" void kernel_launch(void* const* d_in, const int* in_sizes, int n_in,
                              void* d_out, int out_size, void* d_ws, size_t ws_size,
                              hipStream_t stream) {
  const float* wave  = (const float*)d_in[0];
  const float* in_w  = (const float*)d_in[1];
  const float* in_b  = (const float*)d_in[2];
  const float* ln1_g = (const float*)d_in[3];
  const float* ln1_b = (const float*)d_in[4];
  const float* wqk   = (const float*)d_in[5];
  const float* wvw   = (const float*)d_in[6];
  const float* wo_w  = (const float*)d_in[7];
  const float* wo_b  = (const float*)d_in[8];
  const float* rot   = (const float*)d_in[9];
  const float* ln2_g = (const float*)d_in[10];
  const float* ln2_b = (const float*)d_in[11];
  const float* w1    = (const float*)d_in[12];
  const float* b1    = (const float*)d_in[13];
  const float* w2    = (const float*)d_in[14];
  const float* b2    = (const float*)d_in[15];
  const float* outw  = (const float*)d_in[16];
  const float* outb  = (const float*)d_in[17];

  float* ws = (float*)d_ws;
  const size_t TD = (size_t)NTOK * DMODEL;          // 1,310,720
  float* x1      = ws;
  float* x2      = x1 + TD;
  float* xn      = x2 + TD;
  float* qk      = xn + TD;
  float* vv      = qk + TD;
  float* ctx     = vv + TD;
  float* orounds = ctx + TD;                        // 64*5120*16 = 5,242,880
  float* hbuf    = orounds;                         // alias: FF temp reuses attn temp
  float* lsebuf  = orounds + (size_t)BHDIM * NHL * DH;
  int*   buckets = (int*)(lsebuf + (size_t)BHDIM * NHL);
  int*   sticker = buckets + (size_t)BHDIM * NHL;

  embed_kernel<<<(NBATCH * LSEQ * DMODEL + 255) / 256, 256, 0, stream>>>(
      wave, in_w, in_b, x1, x2);

  for (int i = 0; i < 4; ++i) {
    // --- attention branch: x1 += attn(LN(x2)) ---
    ln_kernel<<<NTOK / 4, 128, 0, stream>>>(x2, ln1_g + i * DMODEL, ln1_b + i * DMODEL, xn);
    gemm_wmma<64, false, false, false><<<dim3(NTOK / 64, 4), 128, 0, stream>>>(
        xn, wqk + (size_t)i * DMODEL * DMODEL, nullptr, nullptr, qk, DMODEL);
    gemm_wmma<64, false, false, false><<<dim3(NTOK / 64, 4), 128, 0, stream>>>(
        xn, wvw + (size_t)i * DMODEL * DMODEL, nullptr, nullptr, vv, DMODEL);
    hash_kernel<<<(BHDIM * LSEQ + 255) / 256, 256, 0, stream>>>(
        qk, rot + (size_t)i * DH * NHASH * (NBKT / 2), buckets);
    sort_kernel<<<BHDIM, 128, 0, stream>>>(buckets, sticker);
    attn_kernel<<<dim3(BHDIM, NCHUNK), 128, 0, stream>>>(qk, vv, sticker, orounds, lsebuf);
    combine_kernel<<<(BHDIM * LSEQ + 255) / 256, 256, 0, stream>>>(orounds, lsebuf, ctx);
    gemm_wmma<64, true, false, true><<<dim3(NTOK / 64, 4), 128, 0, stream>>>(
        ctx, wo_w + (size_t)i * DMODEL * DMODEL, wo_b + i * DMODEL, x1, x1, DMODEL);

    // --- FF branch: x2 += ff(LN(x1)) ---
    ln_kernel<<<NTOK / 4, 128, 0, stream>>>(x1, ln2_g + i * DMODEL, ln2_b + i * DMODEL, xn);
    gemm_wmma<64, true, true, false><<<dim3(NTOK / 64, FFDIM / 16), 128, 0, stream>>>(
        xn, w1 + (size_t)i * DMODEL * FFDIM, b1 + i * FFDIM, nullptr, hbuf, FFDIM);
    gemm_wmma<256, true, false, true><<<dim3(NTOK / 64, 4), 128, 0, stream>>>(
        hbuf, w2 + (size_t)i * FFDIM * DMODEL, b2 + i * DMODEL, x2, x2, DMODEL);
  }

  final_kernel<<<(NBATCH * 1250 + 255) / 256, 256, 0, stream>>>(
      x1, x2, outw, outb, (float*)d_out);
}
